// AttnBlock_7954279432551
// MI455X (gfx1250) — matrix operand order repaired and checked by reading, not compiled, not run on hardware
//
#include <hip/hip_runtime.h>
#include <hip/hip_bf16.h>

// ---------------- problem constants ----------------
constexpr int B_   = 16;
constexpr int CIN  = 512;
constexpr int L_   = 1024;
constexpr int LP   = L_ + 8;   // x^T padded rows (4 zero rows each side)
constexpr int COUT = 1024;
constexpr int KW   = 5;        // conv taps
constexpr int DW   = 512;
constexpr int HW   = 196;
constexpr int PSC  = 208;      // HW padded to multiple of 16 (score N dim)
constexpr int PPAD = 224;      // HW padded to multiple of 32 (ctx K dim)
constexpr int CH   = 512;      // GLU output channels (= CIN = DW)

typedef __bf16 bf16_t;
typedef __attribute__((ext_vector_type(16))) __bf16 v16bf;
typedef __attribute__((ext_vector_type(8)))  __bf16 v8bf;
typedef __attribute__((ext_vector_type(8)))  float  v8f;

// ---------------- WMMA helpers (CDNA5 gfx1250, wave32) ----------------
__device__ __forceinline__ v8f wmma_bf16(v16bf a, v16bf b, v8f c) {
  return __builtin_amdgcn_wmma_f32_16x16x32_bf16(
      false, a, false, b, (short)0, c, false, false);
}

// A fragment 16x32 bf16 from a K-contiguous row (row = M row of this lane, lane&15).
// elems 0..7 -> K = half*8 + e ; elems 8..15 -> K = 16 + half*8 + (e-8)
// => two contiguous 16-byte chunks -> 2x global_load_b128.
__device__ __forceinline__ v16bf frag_a16(const bf16_t* __restrict__ row, int half) {
  v8bf lo = *(const v8bf*)(row + (half << 3));
  v8bf hi = *(const v8bf*)(row + 16 + (half << 3));
  return __builtin_shufflevector(lo, hi, 0, 1, 2, 3, 4, 5, 6, 7,
                                 8, 9, 10, 11, 12, 13, 14, 15);
}

// B fragment 32x16 bf16 from B^T stored N-major/K-contiguous
// (row = this lane's column n of B). elems 0..7 -> K = half*8 + e ; elems 8..15 -> K = 16 + half*8 + (e-8)
// => two contiguous 16-byte chunks -> 2x global_load_b128.
__device__ __forceinline__ v16bf frag_b16(const bf16_t* __restrict__ row, int half) {
  v8bf lo = *(const v8bf*)(row + (half << 3));
  v8bf hi = *(const v8bf*)(row + 16 + (half << 3));
  return __builtin_shufflevector(lo, hi, 0, 1, 2, 3, 4, 5, 6, 7,
                                 8, 9, 10, 11, 12, 13, 14, 15);
}

// ---------------- prep kernels ----------------

// Weight-norm: wbf[k][o][i] = g[o]/||v[o]|| * v[o,i,k]  (bf16, K-contiguous in i)
__global__ void k_wnorm(const float* __restrict__ conv_v,
                        const float* __restrict__ conv_g,
                        bf16_t* __restrict__ wbf) {
  const int o = blockIdx.x;
  const int tid = threadIdx.x;
  __shared__ float red[256];
  const float* v = conv_v + (size_t)o * CIN * KW;
  float s = 0.f;
  for (int i = tid; i < CIN * KW; i += 256) { float t = v[i]; s += t * t; }
  red[tid] = s;
  __syncthreads();
  for (int st = 128; st > 0; st >>= 1) {
    if (tid < st) red[tid] += red[tid + st];
    __syncthreads();
  }
  const float scale = conv_g[o] * rsqrtf(red[0]);
  for (int i = tid; i < CIN * KW; i += 256) {
    int ci = i / KW, kk = i % KW;
    wbf[((size_t)kk * COUT + o) * CIN + ci] = (bf16_t)(v[i] * scale);
  }
}

// Tiled transpose x (B,CIN,L) f32 -> xT (B, LP, CIN) bf16 at row offset +4.
__global__ void k_transpose_x(const float* __restrict__ x, bf16_t* __restrict__ xT) {
  __shared__ bf16_t tile[32][33];
  const int b = blockIdx.z;
  const int l0 = blockIdx.x << 5, c0 = blockIdx.y << 5;
  const int tx = threadIdx.x, ty = threadIdx.y; // (32,8)
  for (int i = ty; i < 32; i += 8)
    tile[i][tx] = (bf16_t)x[((size_t)b * CIN + c0 + i) * L_ + l0 + tx];
  __syncthreads();
  for (int i = ty; i < 32; i += 8)
    xT[((size_t)b * LP + 4 + l0 + i) * CIN + c0 + tx] = tile[tx][i];
}

// Zero the 4 leading + 4 trailing pad rows of xT.
__global__ void k_pad_x(bf16_t* __restrict__ xT) {
  int i = blockIdx.x * 256 + threadIdx.x;           // B*8*CIN
  if (i >= B_ * 8 * CIN) return;
  int c = i % CIN, r = (i / CIN) % 8, b = i / (CIN * 8);
  int row = (r < 4) ? r : (LP - 8 + r);
  xT[((size_t)b * LP + row) * CIN + c] = (bf16_t)0.0f;
}

// Plain bf16 casts of fc1_w [a][c] and fc2_w [c][a] (both already K-contiguous).
__global__ void k_cast_fc(const float* __restrict__ fc1, const float* __restrict__ fc2,
                          bf16_t* __restrict__ fc1bf, bf16_t* __restrict__ fc2bf) {
  int i = blockIdx.x * 256 + threadIdx.x;
  if (i >= DW * DW) return;
  fc1bf[i] = (bf16_t)fc1[i];
  fc2bf[i] = (bf16_t)fc2[i];
}

// featpad[b][a][PPAD] (p zero-padded) and featPT[b][p<PSC][a] (transposed).
__global__ void k_prep_feat(const float* __restrict__ img,
                            bf16_t* __restrict__ featpad, bf16_t* __restrict__ featPT) {
  int i = blockIdx.x * 256 + threadIdx.x;
  if (i >= B_ * DW * PPAD) return;
  int p = i % PPAD, a = (i / PPAD) % DW, b = i / (PPAD * DW);
  float v = (p < HW) ? img[((size_t)b * DW + a) * HW + p] : 0.0f;
  featpad[((size_t)b * DW + a) * PPAD + p] = (bf16_t)v;
  if (p < PSC) featPT[((size_t)b * PSC + p) * DW + a] = (bf16_t)v;
}

// ---------------- conv + GLU (dominant GEMM, ~86 GFLOP) ----------------
// y[b,o,t] = sum_k sum_i w[k][o][i] * x[b,i,t+k-4]; h = y_a * sigmoid(y_g)
// grid (L/256, CH/16, B), block (32,4): wave = 16 (a+gate) channels x 64 t.
__global__ void k_conv_glu(const bf16_t* __restrict__ wbf, const bf16_t* __restrict__ xT,
                           const float* __restrict__ conv_b,
                           float* __restrict__ hf, bf16_t* __restrict__ hT) {
  const int lane = threadIdx.x;
  const int half = lane >> 4;
  const int n    = lane & 15;
  const int wave = threadIdx.y;
  const int b    = blockIdx.z;
  const int c0   = blockIdx.y << 4;
  const int t0   = (blockIdx.x << 8) + (wave << 6);

  v8f acc_a[4] = {}; v8f acc_g[4] = {};
  const bf16_t* xb = xT + (size_t)b * LP * CIN;

  for (int k = 0; k < KW; ++k) {
    const bf16_t* wa_row = wbf + ((size_t)k * COUT + (c0 + n)) * CIN;
    const bf16_t* wg_row = wa_row + (size_t)CH * CIN;
    // per-lane B rows: t = t0 + 16*j + n, padded row index tt+4 = t + k
    const bf16_t* xrow0 = xb + ((size_t)(t0 + n + k)) * CIN;
    for (int cb = 0; cb < CIN; cb += 32) {
      v16bf Aa = frag_a16(wa_row + cb, half);
      v16bf Ag = frag_a16(wg_row + cb, half);
#pragma unroll
      for (int j = 0; j < 4; ++j) {
        v16bf Bx = frag_b16(xrow0 + (size_t)(j * 16) * CIN + cb, half);
        acc_a[j] = wmma_bf16(Aa, Bx, acc_a[j]);
        acc_g[j] = wmma_bf16(Ag, Bx, acc_g[j]);
      }
    }
  }
  // epilogue: h = a * sigmoid(g); write hT[b][t][c] (16B chunk) + hf[b][c][l]
  const float ba = conv_b[c0 + (half << 3) + 0];  // per-row biases loaded below
#pragma unroll
  for (int j = 0; j < 4; ++j) {
    const int t = t0 + j * 16 + n;
    v8bf hv;
    float hs[8];
#pragma unroll
    for (int r = 0; r < 8; ++r) {
      const int c = c0 + r + (half << 3);
      const float av = acc_a[j][r] + conv_b[c];
      const float gv = acc_g[j][r] + conv_b[c + CH];
      const float h  = av * (1.0f / (1.0f + __expf(-gv)));
      hs[r] = h;
      hv[r] = (bf16_t)h;
    }
    *(v8bf*)(hT + ((size_t)b * L_ + t) * CIN + c0 + (half << 3)) = hv;
#pragma unroll
    for (int r = 0; r < 8; ++r) {
      const int c = c0 + r + (half << 3);
      hf[((size_t)b * CH + c) * L_ + t] = hs[r];
    }
  }
  (void)ba;
}

// ---------------- q = h^T * fc1^T + b + word_embed -> bf16 ----------------
// grid (DW/256, L/32, B), block (32,4): wave = 32 l x 64 a.
__global__ void k_qgemm(const bf16_t* __restrict__ hT, const bf16_t* __restrict__ fc1bf,
                        const float* __restrict__ fc1_b, const float* __restrict__ wemb,
                        bf16_t* __restrict__ qbf) {
  const int lane = threadIdx.x, half = lane >> 4, n = lane & 15;
  const int wave = threadIdx.y;
  const int b  = blockIdx.z;
  const int l0 = blockIdx.y << 5;
  const int a0 = (blockIdx.x << 8) + (wave << 6);

  v8f acc[2][4] = {};
  const bf16_t* arow0 = hT + ((size_t)b * L_ + l0 + n) * CIN;           // A rows (l)
  const bf16_t* brow0 = fc1bf + ((size_t)(a0 + n)) * DW;                // B rows (a)
  for (int cb = 0; cb < CIN; cb += 32) {
    v16bf A0 = frag_a16(arow0 + cb, half);
    v16bf A1 = frag_a16(arow0 + (size_t)16 * CIN + cb, half);
#pragma unroll
    for (int j = 0; j < 4; ++j) {
      v16bf Bv = frag_b16(brow0 + (size_t)(j * 16) * DW + cb, half);
      acc[0][j] = wmma_bf16(A0, Bv, acc[0][j]);
      acc[1][j] = wmma_bf16(A1, Bv, acc[1][j]);
    }
  }
#pragma unroll
  for (int i = 0; i < 2; ++i)
#pragma unroll
    for (int j = 0; j < 4; ++j)
#pragma unroll
      for (int r = 0; r < 8; ++r) {
        const int l = l0 + 16 * i + r + (half << 3);
        const int a = a0 + 16 * j + n;
        const size_t idx = ((size_t)b * L_ + l) * DW + a;
        qbf[idx] = (bf16_t)(acc[i][j][r] + fc1_b[a] + wemb[idx]);
      }
}

// ---------------- score = q * feat, fused softmax ----------------
// grid (L/16, B), block 32; 13 N-tiles cover PSC=208.
__global__ void k_score_softmax(const bf16_t* __restrict__ qbf,
                                const bf16_t* __restrict__ featPT,
                                float* __restrict__ attn_out,
                                bf16_t* __restrict__ attn_pad) {
  __shared__ float smem[16][PSC];
  const int lane = threadIdx.x, half = lane >> 4, n = lane & 15;
  const int b  = blockIdx.y;
  const int l0 = blockIdx.x << 4;

  v8f acc[13] = {};
  const bf16_t* qrow = qbf + ((size_t)b * L_ + l0 + n) * DW;            // A rows (l)
  const bf16_t* frow = featPT + ((size_t)b * PSC + n) * DW;             // B rows (p)
  for (int ab = 0; ab < DW; ab += 32) {
    v16bf A = frag_a16(qrow + ab, half);
#pragma unroll
    for (int j = 0; j < 13; ++j) {
      v16bf Bv = frag_b16(frow + (size_t)(j * 16) * DW + ab, half);
      acc[j] = wmma_bf16(A, Bv, acc[j]);
    }
  }
#pragma unroll
  for (int j = 0; j < 13; ++j)
#pragma unroll
    for (int r = 0; r < 8; ++r)
      smem[r + (half << 3)][j * 16 + n] = acc[j][r];
  __syncthreads();

  if (lane < 16) {
    const int row = lane;
    const int l = l0 + row;
    float mx = -1e30f;
    for (int p = 0; p < HW; ++p) mx = fmaxf(mx, smem[row][p]);
    float s = 0.f;
    for (int p = 0; p < HW; ++p) { float e = __expf(smem[row][p] - mx); smem[row][p] = e; s += e; }
    const float inv = 1.0f / s;
    bf16_t* ap = attn_pad + ((size_t)b * L_ + l) * PPAD;
    float*  ao = attn_out + ((size_t)b * L_ + l) * HW;
    for (int p = 0; p < HW; ++p) {
      const float a = smem[row][p] * inv;
      ao[p] = a;
      ap[p] = (bf16_t)a;
    }
    for (int p = HW; p < PPAD; ++p) ap[p] = (bf16_t)0.0f;
  }
}

// ---------------- ctx = attn * feat^T -> bf16 ----------------
// grid (DW/256, L/32, B), block (32,4): wave = 32 l x 64 a; K = PPAD.
__global__ void k_ctx(const bf16_t* __restrict__ attn_pad, const bf16_t* __restrict__ featpad,
                      bf16_t* __restrict__ ctxbf) {
  const int lane = threadIdx.x, half = lane >> 4, n = lane & 15;
  const int wave = threadIdx.y;
  const int b  = blockIdx.z;
  const int l0 = blockIdx.y << 5;
  const int a0 = (blockIdx.x << 8) + (wave << 6);

  v8f acc[2][4] = {};
  const bf16_t* arow0 = attn_pad + ((size_t)b * L_ + l0 + n) * PPAD;    // A rows (l)
  const bf16_t* brow0 = featpad + ((size_t)b * DW + a0 + n) * PPAD;     // B rows (a)
  for (int pb = 0; pb < PPAD; pb += 32) {
    v16bf A0 = frag_a16(arow0 + pb, half);
    v16bf A1 = frag_a16(arow0 + (size_t)16 * PPAD + pb, half);
#pragma unroll
    for (int j = 0; j < 4; ++j) {
      v16bf Bv = frag_b16(brow0 + (size_t)(j * 16) * PPAD + pb, half);
      acc[0][j] = wmma_bf16(A0, Bv, acc[0][j]);
      acc[1][j] = wmma_bf16(A1, Bv, acc[1][j]);
    }
  }
#pragma unroll
  for (int i = 0; i < 2; ++i)
#pragma unroll
    for (int j = 0; j < 4; ++j)
#pragma unroll
      for (int r = 0; r < 8; ++r) {
        const int l = l0 + 16 * i + r + (half << 3);
        const int a = a0 + 16 * j + n;
        ctxbf[((size_t)b * L_ + l) * DW + a] = (bf16_t)acc[i][j][r];
      }
}

// ---------------- out[b][c][l] = fc2 * ctx^T + fc2_b + h + x ----------------
// Oriented M=c, N=l so stores/residual reads are lane-coalesced along l.
// grid (L/256, CH/32, B), block (32,4): wave = 32 c x 64 l.
__global__ void k_out(const bf16_t* __restrict__ ctxbf, const bf16_t* __restrict__ fc2bf,
                      const float* __restrict__ fc2_b, const float* __restrict__ hf,
                      const float* __restrict__ x, float* __restrict__ out) {
  const int lane = threadIdx.x, half = lane >> 4, n = lane & 15;
  const int wave = threadIdx.y;
  const int b  = blockIdx.z;
  const int c0 = blockIdx.y << 5;
  const int l0 = (blockIdx.x << 8) + (wave << 6);

  v8f acc[2][4] = {};
  const bf16_t* arow0 = fc2bf + ((size_t)(c0 + n)) * DW;                // A rows (c)
  const bf16_t* brow0 = ctxbf + ((size_t)b * L_ + l0 + n) * DW;         // B rows (l)
  for (int ab = 0; ab < DW; ab += 32) {
    v16bf A0 = frag_a16(arow0 + ab, half);
    v16bf A1 = frag_a16(arow0 + (size_t)16 * DW + ab, half);
#pragma unroll
    for (int j = 0; j < 4; ++j) {
      v16bf Bv = frag_b16(brow0 + (size_t)(j * 16) * DW + ab, half);
      acc[0][j] = wmma_bf16(A0, Bv, acc[0][j]);
      acc[1][j] = wmma_bf16(A1, Bv, acc[1][j]);
    }
  }
#pragma unroll
  for (int i = 0; i < 2; ++i)
#pragma unroll
    for (int j = 0; j < 4; ++j)
#pragma unroll
      for (int r = 0; r < 8; ++r) {
        const int c = c0 + 16 * i + r + (half << 3);
        const int l = l0 + 16 * j + n;
        const size_t oi = ((size_t)b * CH + c) * L_ + l;
        out[oi] = acc[i][j][r] + fc2_b[c] + hf[oi] + x[oi];
      }
}

// ---------------- launch ----------------
extern "C" void kernel_launch(void* const* d_in, const int* in_sizes, int n_in,
                              void* d_out, int out_size, void* d_ws, size_t ws_size,
                              hipStream_t stream) {
  const float* x      = (const float*)d_in[0];
  const float* wemb   = (const float*)d_in[1];
  const float* img    = (const float*)d_in[2];
  // d_in[3] = prev_attn (unused by reference math)
  const float* conv_v = (const float*)d_in[4];
  const float* conv_g = (const float*)d_in[5];
  const float* conv_b = (const float*)d_in[6];
  const float* fc1_w  = (const float*)d_in[7];
  const float* fc1_b  = (const float*)d_in[8];
  const float* fc2_w  = (const float*)d_in[9];
  const float* fc2_b  = (const float*)d_in[10];

  float* out      = (float*)d_out;
  float* out_wemb = out + (size_t)B_ * CH * L_;
  float* out_img  = out_wemb + (size_t)B_ * L_ * DW;
  float* out_attn = out_img + (size_t)B_ * DW * HW;

  // workspace carve (256B aligned slabs)
  char* ws = (char*)d_ws;
  size_t off = 0;
  auto carve = [&](size_t bytes) {
    void* p = ws + off;
    off += (bytes + 255) & ~(size_t)255;
    return p;
  };
  bf16_t* wbf      = (bf16_t*)carve((size_t)KW * COUT * CIN * 2);
  bf16_t* xT       = (bf16_t*)carve((size_t)B_ * LP * CIN * 2);
  bf16_t* fc1bf    = (bf16_t*)carve((size_t)DW * DW * 2);
  bf16_t* fc2bf    = (bf16_t*)carve((size_t)DW * DW * 2);
  bf16_t* featpad  = (bf16_t*)carve((size_t)B_ * DW * PPAD * 2);
  bf16_t* featPT   = (bf16_t*)carve((size_t)B_ * PSC * DW * 2);
  float*  hf       = (float*)carve((size_t)B_ * CH * L_ * 4);
  bf16_t* hT       = (bf16_t*)carve((size_t)B_ * L_ * CIN * 2);
  bf16_t* qbf      = (bf16_t*)carve((size_t)B_ * L_ * DW * 2);
  bf16_t* attn_pad = (bf16_t*)carve((size_t)B_ * L_ * PPAD * 2);
  bf16_t* ctxbf    = (bf16_t*)carve((size_t)B_ * L_ * DW * 2);
  (void)ws_size; (void)in_sizes; (void)n_in; (void)out_size;

  // passthrough outputs
  hipMemcpyAsync(out_wemb, wemb, (size_t)B_ * L_ * DW * 4, hipMemcpyDeviceToDevice, stream);
  hipMemcpyAsync(out_img, img, (size_t)B_ * DW * HW * 4, hipMemcpyDeviceToDevice, stream);

  // prep
  k_wnorm<<<COUT, 256, 0, stream>>>(conv_v, conv_g, wbf);
  k_transpose_x<<<dim3(L_ / 32, CIN / 32, B_), dim3(32, 8), 0, stream>>>(x, xT);
  k_pad_x<<<(B_ * 8 * CIN + 255) / 256, 256, 0, stream>>>(xT);
  k_cast_fc<<<(DW * DW + 255) / 256, 256, 0, stream>>>(fc1_w, fc2_w, fc1bf, fc2bf);
  k_prep_feat<<<(B_ * DW * PPAD + 255) / 256, 256, 0, stream>>>(img, featpad, featPT);

  // main pipeline (all bf16 WMMA, f32 accumulate, 128-bit fragment loads)
  k_conv_glu<<<dim3(L_ / 256, CH / 16, B_), dim3(32, 4), 0, stream>>>(wbf, xT, conv_b, hf, hT);
  k_qgemm<<<dim3(DW / 256, L_ / 32, B_), dim3(32, 4), 0, stream>>>(hT, fc1bf, fc1_b, wemb, qbf);
  k_score_softmax<<<dim3(L_ / 16, B_), 32, 0, stream>>>(qbf, featPT, out_attn, attn_pad);
  k_ctx<<<dim3(DW / 256, L_ / 32, B_), dim3(32, 4), 0, stream>>>(attn_pad, featpad, ctxbf);
  k_out<<<dim3(L_ / 256, CH / 32, B_), dim3(32, 4), 0, stream>>>(ctxbf, fc2bf, fc2_b, hf, x, out);
}
